// OrderedFeatureGroupANFIS_80985903333982
// MI455X (gfx1250) — matrix operand order repaired and checked by reading, not compiled, not run on hardware
//
#include <hip/hip_runtime.h>

typedef __attribute__((ext_vector_type(16))) __bf16 v16bf;
typedef __attribute__((ext_vector_type(8)))  float  v8f;
typedef __attribute__((ext_vector_type(8)))  unsigned int uint8v;

#define BATCH   4096
#define NF      12
#define NM      2
#define NR      4096
#define OD      8
#define TILE_B  16
#define NCOLS_PAD 112                 // 96 (p) + 8 (r) padded to 7*16
#define NTILES  7
#define NCHUNK  128                   // 4096 rules / K=32
// ws layout: [tile 0..6][chunk 0..127][lane 0..31] x 32 bytes (16 bf16 = one B fragment row)
#define WS_BYTES ((size_t)NTILES * NCHUNK * 32 * 32)   // 917504

__device__ __forceinline__ float softplusf(float v) {
    return (v > 20.0f) ? v : log1pf(__expf(v));
}

// ---------------------------------------------------------------------------
// Prep: build P_aug = [consequent_p (R,96) | consequent_r (R,8) | zero-pad]
// pre-swizzled into a B-matrix (32x16 bf16) VGPR layout with contiguous K, same order as A:
//   lane l: col = 16*t + (l&15); lanes 0-15 -> K=0..15, lanes 16-31 -> K=16..31
//   packed element 2v+e  <->  K = (l>>4)*16 + 2v + e,  rule r = 32*chunk + K
// Each lane's fragment is 32 contiguous bytes -> the GEMM loads it with b128s.
// ---------------------------------------------------------------------------
__global__ __launch_bounds__(256)
void prep_bfrag_kernel(const float* __restrict__ cp,
                       const float* __restrict__ cr,
                       uint8v* __restrict__ ws)
{
    const int tid  = blockIdx.x * 256 + threadIdx.x;   // 0 .. 28671
    const int lane = tid & 31;
    const int kkb  = (tid >> 5) & (NCHUNK - 1);
    const int t    = tid >> 12;                        // 0..6
    const int n    = lane & 15;
    const int half = lane >> 4;
    const int col  = t * 16 + n;

    struct Pack { unsigned short s[16]; } pack;
#pragma unroll
    for (int v = 0; v < 8; ++v) {
#pragma unroll
        for (int e = 0; e < 2; ++e) {
            const int K = half * 16 + 2 * v + e;
            const int r = kkb * 32 + K;
            float f;
            if (col < 96)       f = cp[r * 96 + col];
            else if (col < 104) f = cr[r * 8 + (col - 96)];
            else                f = 0.0f;
            pack.s[2 * v + e] = __builtin_bit_cast(unsigned short, (__bf16)f);
        }
    }
    ws[tid] = __builtin_bit_cast(uint8v, pack);        // 32B coalesced store
}

// ---------------------------------------------------------------------------
// Main kernel: out = (firing_norm @ P_aug) combined with x.
// firing[b,r] = low[b, r&63] * high[b, r>>6]  (Kronecker factorization),
// row-sum factorizes too, so normalization folds into low before bf16.
// ---------------------------------------------------------------------------
template<bool USE_WS>
__global__ __launch_bounds__(256)
void anfis_wmma_kernel(const float* __restrict__ x,
                       const float* __restrict__ center_base,
                       const float* __restrict__ center_delta_raw,
                       const float* __restrict__ width_raw,
                       const float* __restrict__ consequent_p,
                       const float* __restrict__ consequent_r,
                       const uint8v* __restrict__ wsB,
                       float* __restrict__ out)
{
    __shared__ float xs[TILE_B][NF];
    __shared__ float mem[TILE_B][NF][NM];
    __shared__ float lowS[TILE_B][64];   // scaled by 1/(sum + 1e-8)
    __shared__ float high[TILE_B][64];
    __shared__ float scale_s[TILE_B];
    __shared__ float cen[NF][NM];
    __shared__ float wid[NF][NM];
    __shared__ float G[TILE_B][NCOLS_PAD];

    const int lane = threadIdx.x;       // 0..31
    const int wave = threadIdx.y;       // 0..7
    const int tid  = wave * 32 + lane;  // 0..255
    const int b0   = blockIdx.x * TILE_B;

    // ---- Phase 1a: ordered centers + widths (softplus) ----
    if (tid < NF) {
        float base = center_base[tid];
        float gap  = softplusf(center_delta_raw[tid]) + 1e-3f;   // (F, M-1=1)
        cen[tid][0] = base;
        cen[tid][1] = base + gap;
        wid[tid][0] = softplusf(width_raw[tid * NM + 0]) + 1e-3f;
        wid[tid][1] = softplusf(width_raw[tid * NM + 1]) + 1e-3f;
    }
    if (tid < TILE_B * NF) {
        int r = tid / NF, f = tid % NF;
        xs[r][f] = x[(b0 + r) * NF + f];
    }
    __syncthreads();

    // ---- Phase 1b: Gaussian memberships ----
    if (tid < TILE_B * NF) {
        int r = tid / NF, f = tid % NF;
        float xv = xs[r][f];
#pragma unroll
        for (int m = 0; m < NM; ++m) {
            float d = xv - cen[f][m];
            float w = wid[f][m];
            mem[r][f][m] = __expf(-(d * d) / (2.0f * w * w));
        }
    }
    __syncthreads();

    // ---- Phase 1c: Kronecker factors of the firing matrix ----
#pragma unroll
    for (int q = 0; q < 4; ++q) {
        int idx = tid * 4 + q;          // 1024 = 16 rows * 64 combos
        int b = idx >> 6, i = idx & 63;
        float pl = 1.0f, ph = 1.0f;
#pragma unroll
        for (int f = 0; f < 6; ++f) {
            int bit = (i >> f) & 1;
            pl *= mem[b][f][bit];
            ph *= mem[b][f + 6][bit];
        }
        lowS[b][i] = pl;
        high[b][i] = ph;
    }
    __syncthreads();

    // ---- Phase 1d: factorized row sum, fold normalization into low ----
    if (tid < TILE_B) {
        float sl = 0.f, sh = 0.f;
#pragma unroll
        for (int i = 0; i < 64; ++i) { sl += lowS[tid][i]; sh += high[tid][i]; }
        scale_s[tid] = 1.0f / (sl * sh + 1e-8f);
    }
    __syncthreads();
#pragma unroll
    for (int q = 0; q < 4; ++q) {
        int idx = tid * 4 + q;
        int b = idx >> 6, i = idx & 63;
        lowS[b][i] *= scale_s[b];
    }
    __syncthreads();

    // ---- Phase 2: G = firing_norm @ P_aug via v_wmma_f32_16x16x32_bf16 ----
    if (wave < NTILES) {
        const int c0t  = wave * 16;
        const int brow = lane & 15;     // A-matrix row M
        const int half = lane >> 4;     // lane half selects K groups

        // Each lane only ever needs 32 lowS values (ibase 0 / 32) -> registers.
        float l0r[16], l32r[16];
#pragma unroll
        for (int v = 0; v < 8; ++v) {
#pragma unroll
            for (int e = 0; e < 2; ++e) {
                int K0 = half * 16 + 2 * v + e;
                l0r [2 * v + e] = lowS[brow][K0];
                l32r[2 * v + e] = lowS[brow][32 + K0];
            }
        }

        // Dual accumulators: breaks the WMMA D->C back-to-back dependency so
        // the compiler doesn't need hazard v_nops between consecutive WMMAs.
        v8f acc0 = {}, acc1 = {};

        if constexpr (USE_WS) {
            // B fragments: one 32B coalesced load per lane per chunk,
            // software-pipelined one iteration ahead (wrap index, branch-free).
            const uint8v* Bw = wsB + (size_t)wave * NCHUNK * 32 + lane;
            uint8v q0 = Bw[0];
            uint8v q1 = Bw[32];
            for (int j = 0; j < 64; ++j) {          // j = high-factor index
                const int jn = (j + 1) & 63;
                uint8v n0 = Bw[(2 * jn + 0) * 32];
                uint8v n1 = Bw[(2 * jn + 1) * 32];
                const float hb = high[brow][j];
                v16bf a0, a1;
#pragma unroll
                for (int i = 0; i < 16; ++i) {
                    a0[i] = (__bf16)(l0r[i]  * hb);
                    a1[i] = (__bf16)(l32r[i] * hb);
                }
                acc0 = __builtin_amdgcn_wmma_f32_16x16x32_bf16(
                    false, a0, false, __builtin_bit_cast(v16bf, q0),
                    (short)0, acc0, false, false);
                acc1 = __builtin_amdgcn_wmma_f32_16x16x32_bf16(
                    false, a1, false, __builtin_bit_cast(v16bf, q1),
                    (short)0, acc1, false, false);
                q0 = n0; q1 = n1;
            }
        } else {
            // Fallback: direct strided f32 gathers of P_aug (no workspace).
            const int ncol = c0t + (lane & 15);
            const float* bptr;  int bstride;  float bmask;
            if (ncol < 96)       { bptr = consequent_p + ncol;        bstride = 96; bmask = 1.f; }
            else if (ncol < 104) { bptr = consequent_r + (ncol - 96); bstride = 8;  bmask = 1.f; }
            else                 { bptr = consequent_p;               bstride = 0;  bmask = 0.f; }
            for (int kk = 0; kk < NR; kk += 64) {
                const float hb = high[brow][kk >> 6];
                v16bf a0, a1, bf0, bf1;
#pragma unroll
                for (int v = 0; v < 8; ++v) {
#pragma unroll
                    for (int e = 0; e < 2; ++e) {
                        a0[2 * v + e] = (__bf16)(l0r [2 * v + e] * hb);
                        a1[2 * v + e] = (__bf16)(l32r[2 * v + e] * hb);
                        int r = kk + half * 16 + 2 * v + e;
                        bf0[2 * v + e] = (__bf16)(bmask * bptr[r * bstride]);
                        bf1[2 * v + e] = (__bf16)(bmask * bptr[(r + 32) * bstride]);
                    }
                }
                acc0 = __builtin_amdgcn_wmma_f32_16x16x32_bf16(
                    false, a0, false, bf0, (short)0, acc0, false, false);
                acc1 = __builtin_amdgcn_wmma_f32_16x16x32_bf16(
                    false, a1, false, bf1, (short)0, acc1, false, false);
            }
        }

        const v8f acc = acc0 + acc1;

        // C/D layout: VGPR p holds row M = p + 8*half, col N = lane&15
#pragma unroll
        for (int p = 0; p < 8; ++p)
            G[p + 8 * half][c0t + (lane & 15)] = acc[p];
    }
    __syncthreads();

    // ---- Phase 3: out[b,o] = sum_f x[b,f]*G[b,f*8+o] + G[b,96+o] ----
    if (tid < TILE_B * OD) {
        int b = tid >> 3, o = tid & 7;
        float val = G[b][96 + o];
#pragma unroll
        for (int f = 0; f < NF; ++f) val += xs[b][f] * G[b][f * 8 + o];
        out[(b0 + b) * OD + o] = val;
    }
}

extern "C" void kernel_launch(void* const* d_in, const int* in_sizes, int n_in,
                              void* d_out, int out_size, void* d_ws, size_t ws_size,
                              hipStream_t stream) {
    const float* x   = (const float*)d_in[0];
    const float* cb  = (const float*)d_in[1];
    const float* cdr = (const float*)d_in[2];
    const float* wr  = (const float*)d_in[3];
    const float* cp  = (const float*)d_in[4];
    const float* cr  = (const float*)d_in[5];
    // d_in[6] (rule_mf_indices) encodes bit f of r for N_MFS=2 -> computed analytically.
    float* outp = (float*)d_out;

    dim3 block(32, 8);                  // 8 wave32 waves
    dim3 grid(BATCH / TILE_B);          // 256 workgroups, 16 batch rows each

    if (ws_size >= WS_BYTES) {
        uint8v* ws = (uint8v*)d_ws;
        hipLaunchKernelGGL(prep_bfrag_kernel, dim3(NTILES * NCHUNK * 32 / 256),
                           dim3(256), 0, stream, cp, cr, ws);
        hipLaunchKernelGGL((anfis_wmma_kernel<true>), grid, block, 0, stream,
                           x, cb, cdr, wr, cp, cr, ws, outp);
    } else {
        hipLaunchKernelGGL((anfis_wmma_kernel<false>), grid, block, 0, stream,
                           x, cb, cdr, wr, cp, cr, (const uint8v*)nullptr, outp);
    }
}
